// WindowAttention_34093450396137
// MI455X (gfx1250) — compile-verified, matrix operand order checked by reading, not run on hardware
//
#include <hip/hip_runtime.h>

// ---------------- problem constants ----------------
#define NTOK   64        // tokens per window (8x8)
#define CDIM   256       // channel dim
#define HEADS  8
#define HDIM   32
#define NBLK   4096
#define SCALEQ 0.17677669529663687f   // 32^-0.5

typedef __attribute__((ext_vector_type(16))) _Float16 v16h;
typedef __attribute__((ext_vector_type(8)))  _Float16 v8h;
typedef __attribute__((ext_vector_type(8)))  float    v8f;
typedef __attribute__((ext_vector_type(4)))  float    v4f;

// ---------------- packed weight panels (in d_ws, f16, WMMA B-layout) ----------
// Tile = 32(K) x 16(N) f16 = 512 halves. Lane l owns halves [l*16, l*16+16):
//   n = l&15, kb = (l>>4)*8 ; halves 0..7 -> K = kb..kb+7, 8..15 -> K = kb+16..kb+23
// Tile order: tile_index = ntile*8 + ktile   (K-major streaming in the GEMM loop)
#define PQ_HALVES  (16 * 8 * 512)      // wq   : 16 ntiles x 8 ktiles
#define PKV_HALVES (32 * 8 * 512)      // wkv  : 32 ntiles x 8 ktiles
#define PP_HALVES  (16 * 8 * 512)      // projw: 16 ntiles x 8 ktiles

// ---------------- LDS layout (byte offsets; strides in halves, bank-spread) --
#define XS    264                      // x tile / attn-out stride  [64][256]+8
#define QKS   536                      // q|k stride                [64][512]+24
#define TS    72                       // V^T stride                [256][64]+8
#define PS    72                       // per-wave P stride         [64][64]+8

#define XH_OFF_B    0                              // 64*264*2      = 33792
#define QK_OFF_B    (XH_OFF_B + 64 * XS * 2)       // +64*536*2     = 68608
#define VT_OFF_B    (QK_OFF_B + 64 * QKS * 2)      // +256*72*2     = 36864
#define PB_OFF_B    (VT_OFF_B + 256 * TS * 2)      // +8*64*72*2    = 73728
#define RIDX_OFF_B  (PB_OFF_B + 8 * 64 * PS * 2)   // +4096 (u8)
#define BT_OFF_B    (RIDX_OFF_B + 64 * 64)         // +1800*4
#define SMEM_BYTES  (BT_OFF_B + 1800 * 4)          // ~219 KB < 320 KB WGP LDS

// ---------------- WMMA helper ----------------
__device__ __forceinline__ v8f wmma16(v16h a, v16h b, v8f c) {
  return __builtin_amdgcn_wmma_f32_16x16x32_f16(false, a, false, b, (short)0, c,
                                                false, false);
}

// A tile (16x32 f16) from a row-major LDS buffer, contiguous along K.
// Also used for B tiles whenever B[k][n] is stored with n as the row and k
// contiguous (k^T reads from QK, V reads from VT).
__device__ __forceinline__ v16h lds_tileA(const _Float16* base, int row0,
                                          int stride, int k0) {
  const int l = threadIdx.x & 31;
  const _Float16* p = base + (row0 + (l & 15)) * stride + k0 + ((l >> 4) << 3);
  v8h lo = *(const v8h*)(p);
  v8h hi = *(const v8h*)(p + 16);
  v16h r;
#pragma unroll
  for (int i = 0; i < 8; ++i) { r[i] = lo[i]; r[i + 8] = hi[i]; }
  return r;
}

// B tile from a pre-swizzled f16 panel: lane l reads 32 contiguous bytes.
__device__ __forceinline__ v16h panel_tileB(const _Float16* __restrict__ tile) {
  const int l = threadIdx.x & 31;
  const v8h* p = (const v8h*)(tile + l * 16);
  v8h lo = p[0], hi = p[1];
  v16h r;
#pragma unroll
  for (int i = 0; i < 8; ++i) { r[i] = lo[i]; r[i + 8] = hi[i]; }
  return r;
}

// Store C/D tile (f32 accum) into an f16 LDS buffer, row-major.
__device__ __forceinline__ void lds_storeC(_Float16* base, int row0, int stride,
                                           int col0, v8f c, float addb) {
  const int l = threadIdx.x & 31;
  const int nn = l & 15, mm = (l >> 4) << 3;
#pragma unroll
  for (int r = 0; r < 8; ++r)
    base[(row0 + mm + r) * stride + col0 + nn] = (_Float16)(c[r] + addb);
}

// Store C/D tile transposed: element (row,col) -> base[col*stride + row].
__device__ __forceinline__ void lds_storeC_T(_Float16* base, int row0,
                                             int stride, int col0, v8f c,
                                             float addb) {
  const int l = threadIdx.x & 31;
  const int nn = l & 15, mm = (l >> 4) << 3;
#pragma unroll
  for (int r = 0; r < 8; ++r)
    base[(col0 + nn) * stride + row0 + mm + r] = (_Float16)(c[r] + addb);
}

// ---------------- pre-pass: f32 weights -> f16 pre-swizzled B panels ---------
// One wave per 32x16 tile; 8 tiles per 256-thread block; 512 tiles total.
__global__ void __launch_bounds__(256)
pack_weights(const float* __restrict__ wq, const float* __restrict__ wkv,
             const float* __restrict__ proj_w, _Float16* __restrict__ pq,
             _Float16* __restrict__ pkv, _Float16* __restrict__ pp) {
  const int tile = blockIdx.x * 8 + (threadIdx.x >> 5);
  const int l = threadIdx.x & 31;
  const float* W; _Float16* P; int ldw, tbase; float mul = 1.0f;
  if (tile < 128)      { W = wq;     P = pq;  ldw = 256; tbase = tile;       mul = SCALEQ; }
  else if (tile < 384) { W = wkv;    P = pkv; ldw = 512; tbase = tile - 128; }
  else                 { W = proj_w; P = pp;  ldw = 256; tbase = tile - 384; }
  const int ntile = tbase >> 3, kt = tbase & 7;
  const int n = l & 15, kb = (l >> 4) << 3;
  const float* src = W + (size_t)(kt * 32 + kb) * ldw + ntile * 16 + n;
  _Float16* dst = P + (size_t)tbase * 512 + l * 16;
  v8h lo, hi;
#pragma unroll
  for (int i = 0; i < 8; ++i) lo[i] = (_Float16)(src[(size_t)i * ldw] * mul);
  src += (size_t)16 * ldw;
#pragma unroll
  for (int i = 0; i < 8; ++i) hi[i] = (_Float16)(src[(size_t)i * ldw] * mul);
  *(v8h*)dst = lo;
  *(v8h*)(dst + 8) = hi;
}

// ---------------- fused window-attention kernel ----------------
__global__ void __launch_bounds__(256, 1)
win_attn_fused(const float* __restrict__ x, const float* __restrict__ bq,
               const float* __restrict__ bkv,
               const float* __restrict__ bias_table,
               const float* __restrict__ proj_b,
               const int* __restrict__ rel_index,
               const _Float16* __restrict__ pq, const _Float16* __restrict__ pkv,
               const _Float16* __restrict__ pp, float* __restrict__ out) {
  extern __shared__ char smem[];
  _Float16* XH  = (_Float16*)(smem + XH_OFF_B);  // x tile f16 [64][XS]; reused as AO
  _Float16* QK  = (_Float16*)(smem + QK_OFF_B);  // q|k f16    [64][QKS]
  _Float16* VT  = (_Float16*)(smem + VT_OFF_B);  // v^T f16    [256][TS]
  _Float16* PB  = (_Float16*)(smem + PB_OFF_B);  // softmax P  [8][64][PS]
  unsigned char* RIDX = (unsigned char*)(smem + RIDX_OFF_B);  // [64*64] u8
  float* BT = (float*)(smem + BT_OFF_B);         // bias table [225*8]

  const int tid  = threadIdx.x;
  const int lane = tid & 31;
  const int wave = tid >> 5;          // 8 wave32
  const int blk  = blockIdx.x;
  const int nlo  = lane & 15;
  const int mb   = (lane >> 4) << 3;

  // ---- stage 0: stage x (f32->f16), rel_index (u8), bias table into LDS ----
  const float* xg = x + (size_t)blk * (NTOK * CDIM);
#pragma unroll
  for (int j = 0; j < 16; ++j) {
    int idx4 = tid + j * 256;                  // 4096 float4 total
    v4f val  = ((const v4f*)xg)[idx4];
    int e    = idx4 * 4;
    _Float16* p = XH + (e >> 8) * XS + (e & 255);
    p[0] = (_Float16)val[0]; p[1] = (_Float16)val[1];
    p[2] = (_Float16)val[2]; p[3] = (_Float16)val[3];
  }
  for (int i = tid; i < 64 * 64; i += 256) RIDX[i] = (unsigned char)rel_index[i];
  for (int i = tid; i < 1800; i += 256) BT[i] = bias_table[i];
  if (tid < 32) {  // warm L2 lines of the freshly packed panels
    __builtin_prefetch(pq + (size_t)tid * 4096, 0, 3);
    __builtin_prefetch(pkv + (size_t)tid * 8192, 0, 3);
    __builtin_prefetch(pp + (size_t)tid * 4096, 0, 3);
  }
  __syncthreads();

  // ---- stage 1: fused QKV projection  [64,256] @ [256,768] -----------------
  // 48 N-tiles of 16 cols; wave w owns tiles [6w, 6w+6). K loop = 8 x 32.
  // q,k go to QK row-major; v goes to VT transposed.
  for (int t = 0; t < 6; ++t) {
    const int j = (wave * 6 + t) * 16;         // output column 0..767
    const _Float16* pan; float bv;
    if (j < 256) { pan = pq  + (size_t)((j >> 4) * 8) * 512;
                   bv  = bq[j + nlo] * SCALEQ; }                 // scale folded
    else         { pan = pkv + (size_t)(((j - 256) >> 4) * 8) * 512;
                   bv  = bkv[j - 256 + nlo]; }
    v8f acc[4] = {v8f{}, v8f{}, v8f{}, v8f{}};
    for (int kt = 0; kt < 8; ++kt) {
      v16h bt = panel_tileB(pan + kt * 512);
#pragma unroll
      for (int mt = 0; mt < 4; ++mt)
        acc[mt] = wmma16(lds_tileA(XH, mt * 16, XS, kt * 32), bt, acc[mt]);
    }
    if (j < 512) {
#pragma unroll
      for (int mt = 0; mt < 4; ++mt)
        lds_storeC(QK, mt * 16, QKS, j, acc[mt], bv);
    } else {
#pragma unroll
      for (int mt = 0; mt < 4; ++mt)
        lds_storeC_T(VT, mt * 16, TS, j - 512, acc[mt], bv);
    }
  }
  __syncthreads();

  // ---- stage 2: attention, one head per wave -------------------------------
  const int h = wave;
  const _Float16* Qh = QK + h * HDIM;          // scale already folded into q
  const _Float16* Kh = QK + 256 + h * HDIM;
  _Float16* Pw = PB + wave * (64 * PS);

  for (int mt = 0; mt < 4; ++mt) {
    // S row-block [16 x 64] = q_tile @ k^T  (K = 32 -> one WMMA per tile)
    v16h aq = lds_tileA(Qh, mt * 16, QKS, 0);
    v8f s[4] = {v8f{}, v8f{}, v8f{}, v8f{}};
#pragma unroll
    for (int nt = 0; nt < 4; ++nt)
      s[nt] = wmma16(aq, lds_tileA(Kh, nt * 16, QKS, 0), s[nt]);

    // bias gather + row softmax (a row's 16 cols live in one VGPR per half)
#pragma unroll
    for (int r = 0; r < 8; ++r) {
      const int m = mt * 16 + mb + r;
      float vals[4];
#pragma unroll
      for (int nt = 0; nt < 4; ++nt) {
        const int n = nt * 16 + nlo;
        vals[nt] = s[nt][r] + BT[(int)RIDX[m * 64 + n] * HEADS + h];
      }
      float mx = fmaxf(fmaxf(vals[0], vals[1]), fmaxf(vals[2], vals[3]));
#pragma unroll
      for (int d = 8; d >= 1; d >>= 1) mx = fmaxf(mx, __shfl_xor(mx, d, 32));
      float e0 = __expf(vals[0] - mx), e1 = __expf(vals[1] - mx);
      float e2 = __expf(vals[2] - mx), e3 = __expf(vals[3] - mx);
      float sum = e0 + e1 + e2 + e3;
#pragma unroll
      for (int d = 8; d >= 1; d >>= 1) sum += __shfl_xor(sum, d, 32);
      const float inv = 1.0f / sum;
      _Float16* pr = Pw + m * PS + nlo;
      pr[0]  = (_Float16)(e0 * inv);
      pr[16] = (_Float16)(e1 * inv);
      pr[32] = (_Float16)(e2 * inv);
      pr[48] = (_Float16)(e3 * inv);
    }
  }

  // out_h = P @ V : [64,64] @ [64,32]; V^T layout makes B tiles contiguous.
  // AO aliases XH (x tile is dead; a barrier separates the phases).
  _Float16* AO = XH;
#pragma unroll
  for (int mt = 0; mt < 4; ++mt) {
#pragma unroll
    for (int dt = 0; dt < 2; ++dt) {
      v8f o = {};
#pragma unroll
      for (int kt = 0; kt < 2; ++kt)
        o = wmma16(lds_tileA(Pw, mt * 16, PS, kt * 32),
                   lds_tileA(VT, h * HDIM + dt * 16, TS, kt * 32), o);
      lds_storeC(AO, mt * 16, XS, h * HDIM + dt * 16, o, 0.0f);
    }
  }
  __syncthreads();

  // ---- stage 3: output projection [64,256] @ [256,256] + bias --------------
  float* og = out + (size_t)blk * (NTOK * CDIM);
  for (int ntl = 0; ntl < 2; ++ntl) {
    const int n0 = (wave * 2 + ntl) * 16;
    const _Float16* pan = pp + (size_t)((n0 >> 4) * 8) * 512;
    v8f acc[4] = {v8f{}, v8f{}, v8f{}, v8f{}};
    for (int kt = 0; kt < 8; ++kt) {
      v16h bt = panel_tileB(pan + kt * 512);
#pragma unroll
      for (int mt = 0; mt < 4; ++mt)
        acc[mt] = wmma16(lds_tileA(AO, mt * 16, XS, kt * 32), bt, acc[mt]);
    }
    const float bv = proj_b[n0 + nlo];
#pragma unroll
    for (int mt = 0; mt < 4; ++mt) {
#pragma unroll
      for (int r = 0; r < 8; ++r)
        og[(mt * 16 + mb + r) * CDIM + n0 + nlo] = acc[mt][r] + bv;
    }
  }
}

// ---------------- host launcher ----------------
extern "C" void kernel_launch(void* const* d_in, const int* in_sizes, int n_in,
                              void* d_out, int out_size, void* d_ws,
                              size_t ws_size, hipStream_t stream) {
  const float* x          = (const float*)d_in[0];
  const float* wq         = (const float*)d_in[1];
  const float* bq         = (const float*)d_in[2];
  const float* wkv        = (const float*)d_in[3];
  const float* bkv        = (const float*)d_in[4];
  const float* bias_table = (const float*)d_in[5];
  const float* proj_w     = (const float*)d_in[6];
  const float* proj_b     = (const float*)d_in[7];
  const int*   rel_index  = (const int*)d_in[8];
  float* out = (float*)d_out;

  (void)in_sizes; (void)n_in; (void)out_size; (void)ws_size;

  // f16 pre-swizzled weight panels in workspace (512 KB total)
  _Float16* pq  = (_Float16*)d_ws;
  _Float16* pkv = pq + PQ_HALVES;
  _Float16* pp  = pkv + PKV_HALVES;

  pack_weights<<<64, 256, 0, stream>>>(wq, wkv, proj_w, pq, pkv, pp);

  hipFuncSetAttribute((const void*)win_attn_fused,
                      hipFuncAttributeMaxDynamicSharedMemorySize, SMEM_BYTES);
  win_attn_fused<<<NBLK, 256, SMEM_BYTES, stream>>>(
      x, bq, bkv, bias_table, proj_b, rel_index, pq, pkv, pp, out);
}
